// PackedTransducerJoiner_29824252903692
// MI455X (gfx1250) — matrix operand order repaired and checked by reading, not compiled, not run on hardware
//
#include <hip/hip_runtime.h>
#include <hip/hip_bf16.h>

// ---------------------------------------------------------------------------
// Fused packed-transducer joiner for gfx1250 (MI455X), bf16 WMMA path.
//   GEMM1: joint[N,1024](bf16) x W1[1024,640](bf16)  -> h (f32 acc, ReLU)
//   GEMM2: h[N,640](bf16)      x W2[640,1024](bf16)  -> logits (f32 acc)
//   log_softmax fused; only the final [N,1024] f32 hits HBM (non-temporal).
// BM=32: each wave owns BOTH 16-row M-tiles so every B fragment (1KB) feeds
// two v_wmma ops -> halves L0/L2 B-traffic per FLOP vs BM=16.
// ---------------------------------------------------------------------------

typedef __attribute__((ext_vector_type(16))) __bf16 v16bf;
typedef __attribute__((ext_vector_type(8)))  float  v8f;

#define D_      512
#define T_      400
#define U_      50
#define K1_     1024   // 2*D
#define LAYER_  640
#define VOCAB_  1024

#define BM      32            // rows of N per workgroup (2 M-tiles)
#define MT      2             // M-tiles per wave
#define NW      8             // waves per workgroup (256 threads, wave32)
#define KT1     (K1_/32)      // 32 k-steps, gemm1
#define KT2     (LAYER_/32)   // 20 k-steps, gemm2
#define NT1     (LAYER_/16)   // 40 n-tiles, gemm1
#define NT2     (VOCAB_/16)   // 64 n-tiles, gemm2
#define NT1W    (NT1/NW)      // 5 n-tiles per wave, gemm1
#define NT2W    (NT2/NW)      // 8 n-tiles per wave, gemm2

#define JROW    1032          // jointS row pitch in halves (bank-friendly)
#define HROW    648           // hS row pitch in halves (overlaid on jointS)

#define W1P_HALVES (K1_*LAYER_)     // 655360
#define W2P_HALVES (LAYER_*VOCAB_)  // 655360

union AFrag { uint4 q[2]; v16bf v; };

__device__ __forceinline__ unsigned short f2bf(float x) {
    unsigned u = __float_as_uint(x);
    u += 0x7FFFu + ((u >> 16) & 1u);          // round-to-nearest-even
    return (unsigned short)(u >> 16);
}
// ordered-float <-> unsigned key (monotonic) for atomic max on shared mem
__device__ __forceinline__ unsigned fenc(float f) {
    unsigned u = __float_as_uint(f);
    return (u & 0x80000000u) ? ~u : (u | 0x80000000u);
}
__device__ __forceinline__ float fdec(unsigned k) {
    unsigned u = (k & 0x80000000u) ? (k & 0x7FFFFFFFu) : ~k;
    return __uint_as_float(u);
}

// ---------------------------------------------------------------------------
// Prep: swizzle W1/W2 (f32, row-major [K,N]) into bf16 B-fragment-linear
// tiles: tile(kt,nt) is 32(K)x16(N); lane l owns column nt*16+(l&15),
// K = kt*32 + (l<16?0:8) + (j&7) + (j<8?0:16), j=0..15 -> 32B per lane.
// ---------------------------------------------------------------------------
__global__ void prep_weights(const float* __restrict__ W1,
                             const float* __restrict__ W2,
                             unsigned short* __restrict__ wsp) {
    int id = blockIdx.x * blockDim.x + threadIdx.x;
    if (id < W1P_HALVES) {
        int tile = id >> 9;          // /512
        int rem  = id & 511;
        int l    = rem >> 4;
        int j    = rem & 15;
        int kt   = tile / NT1, nt = tile % NT1;
        int K    = kt * 32 + ((l & 16) ? 8 : 0) + (j & 7) + ((j & 8) ? 16 : 0);
        int C    = nt * 16 + (l & 15);
        wsp[id]  = f2bf(W1[K * LAYER_ + C]);
    } else if (id < W1P_HALVES + W2P_HALVES) {
        int id2  = id - W1P_HALVES;
        int tile = id2 >> 9;
        int rem  = id2 & 511;
        int l    = rem >> 4;
        int j    = rem & 15;
        int kt   = tile / NT2, nt = tile % NT2;
        int K    = kt * 32 + ((l & 16) ? 8 : 0) + (j & 7) + ((j & 8) ? 16 : 0);
        int C    = nt * 16 + (l & 15);
        wsp[id]  = f2bf(W2[K * VOCAB_ + C]);
    }
}

// ---------------------------------------------------------------------------
// Main fused kernel: one workgroup per 32 packed rows.
// ---------------------------------------------------------------------------
__global__ void __launch_bounds__(256)
joiner_kernel(const float* __restrict__ srcEnc, const float* __restrict__ tgtEnc,
              const int* __restrict__ b_idx, const int* __restrict__ t_idx,
              const int* __restrict__ u_idx,
              const unsigned short* __restrict__ W1p,
              const unsigned short* __restrict__ W2p,
              const float* __restrict__ b1, const float* __restrict__ b2,
              float* __restrict__ out, int N) {
    // jointS[BM][JROW] overlaid with hS[BM][HROW] (barrier-separated phases)
    __shared__ __align__(16) unsigned short smem[BM * JROW];
    __shared__ unsigned rowMaxK[BM];
    __shared__ float    rowSum[BM];

    const int tid  = threadIdx.x;
    const int rowBase = blockIdx.x * BM;

    // ---- phase 0: gather + concat rows into LDS as bf16 -------------------
    {
        int i   = tid >> 3;        // row 0..31
        int sub = tid & 7;         // 8 threads per row
        int ridx = rowBase + i; if (ridx >= N) ridx = 0;
        int b = b_idx[ridx], t = t_idx[ridx], u = u_idx[ridx];
        const float4* srow = (const float4*)(srcEnc + ((size_t)b * T_ + t) * D_);
        const float4* trow = (const float4*)(tgtEnc + ((size_t)b * U_ + u) * D_);
        unsigned short* jrow = smem + i * JROW;
        #pragma unroll
        for (int c = 0; c < (D_ / 4) / 8; ++c) {       // 16 float4 per thread/half
            int k4 = sub + c * 8;
            float4 s = srow[k4];
            float4 g = trow[k4];
            uint2 ps, pg;
            ps.x = (unsigned)f2bf(s.x) | ((unsigned)f2bf(s.y) << 16);
            ps.y = (unsigned)f2bf(s.z) | ((unsigned)f2bf(s.w) << 16);
            pg.x = (unsigned)f2bf(g.x) | ((unsigned)f2bf(g.y) << 16);
            pg.y = (unsigned)f2bf(g.z) | ((unsigned)f2bf(g.w) << 16);
            *(uint2*)&jrow[k4 * 4]       = ps;
            *(uint2*)&jrow[D_ + k4 * 4]  = pg;
        }
    }
    __syncthreads();

    const int w      = tid >> 5;
    const int lane   = tid & 31;
    const int mrow   = lane & 15;
    const int lane15 = lane & 15;
    const int lhalf  = (lane & 16) ? 8 : 0;  // A/C layout split at lane 16

    // ---- phase 1: GEMM1 (joint @ W1), K=1024 ------------------------------
    v8f acc[MT][NT1W] = {};
    for (int kt = 0; kt < KT1; ++kt) {
        AFrag a0, a1;   // both M-tiles share each B fragment
        a0.q[0] = *(const uint4*)&smem[(mrow)      * JROW + kt * 32 + lhalf];
        a0.q[1] = *(const uint4*)&smem[(mrow)      * JROW + kt * 32 + 16 + lhalf];
        a1.q[0] = *(const uint4*)&smem[(mrow + 16) * JROW + kt * 32 + lhalf];
        a1.q[1] = *(const uint4*)&smem[(mrow + 16) * JROW + kt * 32 + 16 + lhalf];
        #pragma unroll
        for (int nt = 0; nt < NT1W; ++nt) {
            int tile = kt * NT1 + w * NT1W + nt;
            const uint4* bp = (const uint4*)(W1p + (size_t)tile * 512 + lane * 16);
            AFrag bf; bf.q[0] = bp[0]; bf.q[1] = bp[1];
            acc[0][nt] = __builtin_amdgcn_wmma_f32_16x16x32_bf16(
                false, a0.v, false, bf.v, (short)0, acc[0][nt], false, false);
            acc[1][nt] = __builtin_amdgcn_wmma_f32_16x16x32_bf16(
                false, a1.v, false, bf.v, (short)0, acc[1][nt], false, false);
        }
    }
    __syncthreads();   // all waves done reading jointS before hS overlay write
    if (tid < BM) { rowMaxK[tid] = 0u; rowSum[tid] = 0.0f; }

    // bias + ReLU -> hS (bf16, overlaid region)
    #pragma unroll
    for (int nt = 0; nt < NT1W; ++nt) {
        int col = w * (NT1W * 16) + nt * 16 + lane15;
        float bias = b1[col];
        #pragma unroll
        for (int mt = 0; mt < MT; ++mt) {
            #pragma unroll
            for (int r = 0; r < 8; ++r) {
                float v = acc[mt][nt][r] + bias;
                v = fmaxf(v, 0.0f);
                smem[(mt * 16 + r + lhalf) * HROW + col] = f2bf(v);
            }
        }
    }
    __syncthreads();

    // ---- phase 2: GEMM2 (h @ W2), K=640 -----------------------------------
    v8f acc2[MT][NT2W] = {};
    for (int kt = 0; kt < KT2; ++kt) {
        AFrag a0, a1;
        a0.q[0] = *(const uint4*)&smem[(mrow)      * HROW + kt * 32 + lhalf];
        a0.q[1] = *(const uint4*)&smem[(mrow)      * HROW + kt * 32 + 16 + lhalf];
        a1.q[0] = *(const uint4*)&smem[(mrow + 16) * HROW + kt * 32 + lhalf];
        a1.q[1] = *(const uint4*)&smem[(mrow + 16) * HROW + kt * 32 + 16 + lhalf];
        #pragma unroll
        for (int nt = 0; nt < NT2W; ++nt) {
            int tile = kt * NT2 + w * NT2W + nt;
            const uint4* bp = (const uint4*)(W2p + (size_t)tile * 512 + lane * 16);
            AFrag bf; bf.q[0] = bp[0]; bf.q[1] = bp[1];
            acc2[0][nt] = __builtin_amdgcn_wmma_f32_16x16x32_bf16(
                false, a0.v, false, bf.v, (short)0, acc2[0][nt], false, false);
            acc2[1][nt] = __builtin_amdgcn_wmma_f32_16x16x32_bf16(
                false, a1.v, false, bf.v, (short)0, acc2[1][nt], false, false);
        }
    }
    // add b2
    #pragma unroll
    for (int nt = 0; nt < NT2W; ++nt) {
        int col = w * (NT2W * 16) + nt * 16 + lane15;
        float bias = b2[col];
        #pragma unroll
        for (int mt = 0; mt < MT; ++mt)
            #pragma unroll
            for (int r = 0; r < 8; ++r) acc2[mt][nt][r] += bias;
    }

    // ---- phase 3: fused log-softmax over VOCAB ----------------------------
    #pragma unroll
    for (int mt = 0; mt < MT; ++mt) {
        #pragma unroll
        for (int r = 0; r < 8; ++r) {
            float mx = -3.4e38f;
            #pragma unroll
            for (int nt = 0; nt < NT2W; ++nt) mx = fmaxf(mx, acc2[mt][nt][r]);
            atomicMax(&rowMaxK[mt * 16 + r + lhalf], fenc(mx));
        }
    }
    __syncthreads();
    #pragma unroll
    for (int mt = 0; mt < MT; ++mt) {
        #pragma unroll
        for (int r = 0; r < 8; ++r) {
            float M = fdec(rowMaxK[mt * 16 + r + lhalf]);
            float s = 0.0f;
            #pragma unroll
            for (int nt = 0; nt < NT2W; ++nt) s += __expf(acc2[mt][nt][r] - M);
            atomicAdd(&rowSum[mt * 16 + r + lhalf], s);
        }
    }
    __syncthreads();
    #pragma unroll
    for (int mt = 0; mt < MT; ++mt) {
        #pragma unroll
        for (int r = 0; r < 8; ++r) {
            int m = mt * 16 + r + lhalf;
            int grow = rowBase + m;
            if (grow < N) {
                float M = fdec(rowMaxK[m]);
                float L = __logf(rowSum[m]);
                size_t base = (size_t)grow * VOCAB_ + w * (NT2W * 16) + lane15;
                #pragma unroll
                for (int nt = 0; nt < NT2W; ++nt)
                    __builtin_nontemporal_store(acc2[mt][nt][r] - M - L,
                                                &out[base + nt * 16]);
            }
        }
    }
}

// tuple tail: (out, source_lengths, target_lengths) concatenated flat
__global__ void write_lengths(const int* __restrict__ sl, const int* __restrict__ tl,
                              float* __restrict__ out, size_t off) {
    int i = threadIdx.x;
    if (i < 8)       out[off + i] = (float)sl[i];
    else if (i < 16) out[off + i] = (float)tl[i - 8];
}

extern "C" void kernel_launch(void* const* d_in, const int* in_sizes, int n_in,
                              void* d_out, int out_size, void* d_ws, size_t ws_size,
                              hipStream_t stream) {
    const float* srcEnc = (const float*)d_in[0];
    const int*   sLen   = (const int*)d_in[1];
    const float* tgtEnc = (const float*)d_in[2];
    const int*   tLen   = (const int*)d_in[3];
    const int*   bI     = (const int*)d_in[4];
    const int*   tI     = (const int*)d_in[5];
    const int*   uI     = (const int*)d_in[6];
    const float* W1     = (const float*)d_in[7];
    const float* b1     = (const float*)d_in[8];
    const float* W2     = (const float*)d_in[9];
    const float* b2     = (const float*)d_in[10];
    float*       out    = (float*)d_out;
    const int N = in_sizes[4];

    unsigned short* W1p = (unsigned short*)d_ws;
    unsigned short* W2p = W1p + W1P_HALVES;

    int prepTotal = W1P_HALVES + W2P_HALVES;
    prep_weights<<<(prepTotal + 255) / 256, 256, 0, stream>>>(W1, W2, W1p);

    int groups = (N + BM - 1) / BM;
    joiner_kernel<<<groups, 256, 0, stream>>>(srcEnc, tgtEnc, bI, tI, uI,
                                              W1p, W2p, b1, b2, out, N);

    write_lengths<<<1, 16, 0, stream>>>(sLen, tLen, out, (size_t)N * VOCAB_);
}
